// DPC3_NET_33861522162110
// MI455X (gfx1250) — matrix operand order repaired and checked by reading, not compiled, not run on hardware
//
#include <hip/hip_runtime.h>
#include <hip/hip_bf16.h>

typedef _Float16 v16h __attribute__((ext_vector_type(16)));
typedef float    v8f  __attribute__((ext_vector_type(8)));

#define BM 64     // block tile M (Cout)
#define BN 128    // block tile N (spatial)
#define TK 32     // K step

#if __has_builtin(__builtin_amdgcn_global_load_async_to_lds_b128)
#define HAVE_ASYNC_LDS 1
typedef int v4i32 __attribute__((vector_size(16)));
typedef __attribute__((address_space(1))) v4i32 gv4i32;
typedef __attribute__((address_space(3))) v4i32 lv4i32;
#else
#define HAVE_ASYNC_LDS 0
#endif

// ---------------------------------------------------------------------------
// Implicit-GEMM conv (3x3 or 1x1), WMMA f16 -> f32 accum, fused folded-BN,
// optional residual + ReLU. Activations NCHW f16; weights [Cout][Cin*k*k] f16.
// Requires Cout % 64 == 0 and (Cin*k*k) % 32 == 0 (true for all ResNet18
// residual-stage convs). 8 wave32s; each wave owns a 32x32 output tile
// (2 A-frags x 2 B-frags -> 4 WMMAs per K step).
// ---------------------------------------------------------------------------
__global__ __launch_bounds__(256)
void conv_wmma_kernel(const _Float16* __restrict__ act,
                      const _Float16* __restrict__ wgt,
                      const float* __restrict__ sc,
                      const float* __restrict__ bi,
                      const _Float16* __restrict__ resid,
                      _Float16* __restrict__ out,
                      int Bn, int Cin, int Hi, int Wi,
                      int Cout, int Ho, int Wo,
                      int KW, int stride, int pad, int doRelu)
{
    __shared__ _Float16 As[BM][TK];   // weights: [m][k]
    __shared__ _Float16 Bs[BN][TK];   // activations, transposed: [n][k]

    const int K    = Cin * KW * KW;
    const int KK   = KW * KW;
    const int HoWo = Ho * Wo;
    const int Ntot = Bn * HoWo;

    const int tid  = threadIdx.x;
    const int lane = tid & 31;
    const int wave = tid >> 5;
    const int g    = (lane >> 4) & 1;   // lane group (0: lanes 0-15, 1: 16-31)
    const int ln   = lane & 15;

    const int m0 = blockIdx.y * BM;
    const int n0 = blockIdx.x * BN;
    const int mw = (wave >> 2) * 32;    // wave M offset within tile (0,32)
    const int nw = (wave & 3) * 32;     // wave N offset within tile (0,32,64,96)

    v8f acc00 = {}, acc01 = {}, acc10 = {}, acc11 = {};

    for (int k0 = 0; k0 < K; k0 += TK) {
        // ---- stage A (weights): 64x32 halves, 16B per thread ----
        {
            const int m  = tid >> 2;          // 0..63
            const int kc = (tid & 3) * 8;     // 0,8,16,24
            const _Float16* src = &wgt[(size_t)(m0 + m) * K + (k0 + kc)];
#if HAVE_ASYNC_LDS
            __builtin_amdgcn_global_load_async_to_lds_b128(
                (gv4i32*)src, (lv4i32*)&As[m][kc], 0, 0);
#else
            *(uint4*)&As[m][kc] = *(const uint4*)src;
#endif
            if (k0 + TK < K) __builtin_prefetch(src + TK, 0, 0);
        }
        // ---- stage B (im2col gather): 128x32 halves ----
        for (int e = tid; e < BN * TK; e += 256) {
            const int tn = e >> 5;
            const int tk = e & 31;
            const int k  = k0 + tk;
            const int ci = k / KK;
            const int rm = k - ci * KK;
            const int kh = rm / KW;
            const int kw = rm - kh * KW;
            _Float16 val = (_Float16)0.f;
            const int n = n0 + tn;
            if (n < Ntot) {
                const int b  = n / HoWo;
                const int r  = n - b * HoWo;
                const int ho = r / Wo;
                const int wo = r - ho * Wo;
                const int hi = ho * stride - pad + kh;
                const int wi = wo * stride - pad + kw;
                if (hi >= 0 && hi < Hi && wi >= 0 && wi < Wi)
                    val = act[(((size_t)b * Cin + ci) * Hi + hi) * Wi + wi];
            }
            Bs[tn][tk] = val;
        }
#if HAVE_ASYNC_LDS
#if __has_builtin(__builtin_amdgcn_s_wait_asynccnt)
        __builtin_amdgcn_s_wait_asynccnt(0);
#else
        asm volatile("s_wait_asynccnt 0x0" ::: "memory");
#endif
#endif
        __syncthreads();

        // ---- load fragments per ISA VGPR layout, issue 4 WMMAs ----
        union { v16h h; unsigned u[8]; } fa0, fa1, fb0, fb1;
#pragma unroll
        for (int v = 0; v < 8; ++v) {
            const int kb  = (v < 4) ? (g * 8 + 2 * v) : (16 + g * 8 + 2 * (v - 4));
            fa0.u[v] = *(const unsigned*)&As[mw + ln][kb];
            fa1.u[v] = *(const unsigned*)&As[mw + 16 + ln][kb];
            const int kb2 = (v < 4) ? (g * 8 + 2 * v) : (16 + g * 8 + 2 * (v - 4));
            fb0.u[v] = *(const unsigned*)&Bs[nw + ln][kb2];
            fb1.u[v] = *(const unsigned*)&Bs[nw + 16 + ln][kb2];
        }
        acc00 = __builtin_amdgcn_wmma_f32_16x16x32_f16(false, fa0.h, false, fb0.h,
                                                       (short)0, acc00, false, false);
        acc01 = __builtin_amdgcn_wmma_f32_16x16x32_f16(false, fa0.h, false, fb1.h,
                                                       (short)0, acc01, false, false);
        acc10 = __builtin_amdgcn_wmma_f32_16x16x32_f16(false, fa1.h, false, fb0.h,
                                                       (short)0, acc10, false, false);
        acc11 = __builtin_amdgcn_wmma_f32_16x16x32_f16(false, fa1.h, false, fb1.h,
                                                       (short)0, acc11, false, false);
        __syncthreads();
    }

    // ---- epilogue: BN + residual + ReLU, write f16 NCHW ----
#pragma unroll
    for (int half = 0; half < 2; ++half) {
#pragma unroll
        for (int r = 0; r < 8; ++r) {
            const int m = m0 + mw + half * 16 + r + 8 * g;   // output channel
            const float s  = sc[m];
            const float bb = bi[m];
#pragma unroll
            for (int sub = 0; sub < 2; ++sub) {
                const int n = n0 + nw + sub * 16 + ln;
                if (n >= Ntot) continue;
                const float av = half ? (sub ? acc11[r] : acc10[r])
                                      : (sub ? acc01[r] : acc00[r]);
                float f = av * s + bb;
                const int b  = n / HoWo;
                const int rr = n - b * HoWo;
                const size_t addr = ((size_t)b * Cout + m) * HoWo + rr;
                if (resid) f += (float)resid[addr];
                if (doRelu) f = fmaxf(f, 0.f);
                out[addr] = (_Float16)f;
            }
        }
    }
}

// ---------------------------------------------------------------------------
// Stem: 7x7/2 pad3 conv (Cin=3) + BN + ReLU, direct form, f32 in -> f16 out.
// ---------------------------------------------------------------------------
__global__ void stem_conv_kernel(const float* __restrict__ x,
                                 const float* __restrict__ w,
                                 const float* __restrict__ sc,
                                 const float* __restrict__ bi,
                                 _Float16* __restrict__ out, int total)
{
    int i = blockIdx.x * blockDim.x + threadIdx.x;
    if (i >= total) return;
    int wo = i % 112; int t = i / 112;
    int ho = t % 112; t /= 112;
    int c  = t % 64;
    int b  = t / 64;
    float acc = 0.f;
    for (int ci = 0; ci < 3; ++ci)
        for (int kh = 0; kh < 7; ++kh) {
            int hi = ho * 2 - 3 + kh;
            if (hi < 0 || hi >= 224) continue;
            for (int kw = 0; kw < 7; ++kw) {
                int wi = wo * 2 - 3 + kw;
                if (wi < 0 || wi >= 224) continue;
                acc += x[(((size_t)b * 3 + ci) * 224 + hi) * 224 + wi]
                     * w[(((size_t)c * 3 + ci) * 7 + kh) * 7 + kw];
            }
        }
    float f = acc * sc[c] + bi[c];
    out[i] = (_Float16)fmaxf(f, 0.f);
}

// 3x3/2 pad1 maxpool, f16 -> f16 (112 -> 56)
__global__ void maxpool_kernel(const _Float16* __restrict__ in,
                               _Float16* __restrict__ out, int total)
{
    int i = blockIdx.x * blockDim.x + threadIdx.x;
    if (i >= total) return;
    int wo = i % 56; int t = i / 56;
    int ho = t % 56; t /= 56;
    int c  = t % 64;
    int b  = t / 64;
    float m = -3.0e38f;
    for (int kh = 0; kh < 3; ++kh) {
        int hi = ho * 2 - 1 + kh;
        if (hi < 0 || hi >= 112) continue;
        for (int kw = 0; kw < 3; ++kw) {
            int wi = wo * 2 - 1 + kw;
            if (wi < 0 || wi >= 112) continue;
            m = fmaxf(m, (float)in[(((size_t)b * 64 + c) * 112 + hi) * 112 + wi]);
        }
    }
    out[i] = (_Float16)m;
}

// global average pool over 7x7, f16 [32][512][7][7] -> f32 [32][512]
__global__ void avgpool_kernel(const _Float16* __restrict__ in,
                               float* __restrict__ out)
{
    int i = blockIdx.x * blockDim.x + threadIdx.x;
    if (i >= 32 * 512) return;
    const _Float16* p = in + (size_t)i * 49;
    float s = 0.f;
    for (int j = 0; j < 49; ++j) s += (float)p[j];
    out[i] = s * (1.0f / 49.0f);
}

// FC1: [32][512] x W[256][512]^T + b -> [32][256]
__global__ void fc1_kernel(const float* __restrict__ v,
                           const float* __restrict__ w,
                           const float* __restrict__ b,
                           float* __restrict__ out)
{
    int i = blockIdx.x * blockDim.x + threadIdx.x;
    if (i >= 32 * 256) return;
    int o  = i % 256;
    int bb = i / 256;
    const float* vr = v + (size_t)bb * 512;
    const float* wr = w + (size_t)o * 512;
    float s = b[o];
    for (int c = 0; c < 512; ++c) s += vr[c] * wr[c];
    out[i] = s;
}

// Decoder: per point feat=[v(256),u,v] -> relu(W1 f + b1) -> tanh(W2 z + b2)
__global__ __launch_bounds__(256)
void decoder_kernel(const float* __restrict__ v256,
                    const float* __restrict__ coords,
                    const float* __restrict__ w1, const float* __restrict__ b1,
                    const float* __restrict__ w2, const float* __restrict__ b2,
                    float* __restrict__ y)
{
    __shared__ float sh[256];
    const int b = blockIdx.y;
    const int n = blockIdx.x * 256 + threadIdx.x;
    sh[threadIdx.x] = v256[(size_t)b * 256 + threadIdx.x];
    __syncthreads();
    if (n >= 2500) return;
    const float u  = coords[((size_t)b * 2500 + n) * 2 + 0];
    const float vv = coords[((size_t)b * 2500 + n) * 2 + 1];
    float y0 = b2[0], y1 = b2[1], y2 = b2[2];
    for (int o = 0; o < 129; ++o) {
        const float* wr = w1 + (size_t)o * 258;
        float z = b1[o];
        for (int c = 0; c < 256; ++c) z += wr[c] * sh[c];
        z += wr[256] * u + wr[257] * vv;
        z = fmaxf(z, 0.f);
        y0 += w2[0 * 129 + o] * z;
        y1 += w2[1 * 129 + o] * z;
        y2 += w2[2 * 129 + o] * z;
    }
    y[((size_t)b * 3 + 0) * 2500 + n] = tanhf(y0);
    y[((size_t)b * 3 + 1) * 2500 + n] = tanhf(y1);
    y[((size_t)b * 3 + 2) * 2500 + n] = tanhf(y2);
}

// f32 -> f16 weight conversion
__global__ void f2h_kernel(const float* __restrict__ s,
                           _Float16* __restrict__ d, int n)
{
    int i = blockIdx.x * blockDim.x + threadIdx.x;
    if (i < n) d[i] = (_Float16)s[i];
}

// fold BN: sc = scale*rsqrt(var+eps), bi = bias - mean*sc
__global__ void bnfold_kernel(const float* __restrict__ scale,
                              const float* __restrict__ bias,
                              const float* __restrict__ mean,
                              const float* __restrict__ var,
                              float* __restrict__ sc, float* __restrict__ bi, int c)
{
    int i = blockIdx.x * blockDim.x + threadIdx.x;
    if (i < c) {
        float inv = rsqrtf(var[i] + 1e-5f) * scale[i];
        sc[i] = inv;
        bi[i] = bias[i] - mean[i] * inv;
    }
}

extern "C" void kernel_launch(void* const* d_in, const int* in_sizes, int n_in,
                              void* d_out, int out_size, void* d_ws, size_t ws_size,
                              hipStream_t stream)
{
    (void)in_sizes; (void)n_in; (void)out_size; (void)ws_size;
    auto F = [&](int i) { return (const float*)d_in[i]; };

    // ---- input parse (jax tree order: dict keys alphabetical) ----
    int p = 0;
    const float* X        = F(p++);
    const float* coords   = F(p++);
    const float* bn1_bias = F(p++); const float* bn1_mean = F(p++);
    const float* bn1_scale= F(p++); const float* bn1_var  = F(p++);
    const float* conv1_w  = F(p++);
    const float* dec1_b   = F(p++); const float* dec1_w   = F(p++);
    const float* dec2_b   = F(p++); const float* dec2_w   = F(p++);
    const float* fc1_b    = F(p++); const float* fc1_w    = F(p++);

    // layer table: 0 = stem; then per block: conv1, conv2, [down]
    const float* Wsrc[21]; const float *Lb[21], *Lm[21], *Ls[21], *Lv[21];
    int Wn[21]; int Co[21];
    int L = 0;
    Wsrc[L]=conv1_w; Lb[L]=bn1_bias; Lm[L]=bn1_mean; Ls[L]=bn1_scale; Lv[L]=bn1_var;
    Wn[L]=64*3*49; Co[L]=64; L++;

    const int cins[4]  = {64, 64, 128, 256};
    const int couts[4] = {64, 128, 256, 512};

    for (int s = 0; s < 4; ++s)
        for (int i = 0; i < 2; ++i) {
            int ci = (i == 0) ? cins[s] : couts[s];
            int co = couts[s];
            const float *b1b=F(p++), *b1m=F(p++), *b1s=F(p++), *b1v=F(p++);
            const float *b2b=F(p++), *b2m=F(p++), *b2s=F(p++), *b2v=F(p++);
            const float *c1 = F(p++);
            const float *c2 = F(p++);
            Wsrc[L]=c1; Lb[L]=b1b; Lm[L]=b1m; Ls[L]=b1s; Lv[L]=b1v; Wn[L]=co*ci*9; Co[L]=co; L++;
            Wsrc[L]=c2; Lb[L]=b2b; Lm[L]=b2m; Ls[L]=b2s; Lv[L]=b2v; Wn[L]=co*co*9; Co[L]=co; L++;
            if (s > 0 && i == 0) {
                const float *dbb=F(p++), *dbm=F(p++), *dbs=F(p++), *dbv=F(p++);
                const float *dw = F(p++);
                Wsrc[L]=dw; Lb[L]=dbb; Lm[L]=dbm; Ls[L]=dbs; Lv[L]=dbv; Wn[L]=co*ci; Co[L]=co; L++;
            }
        }
    const int nLayers = L;   // 20

    // ---- workspace carving ----
    size_t off = 0;
    auto carve = [&](size_t bytes) -> void* {
        void* r = (void*)((char*)d_ws + off);
        off += (bytes + 255) & ~(size_t)255;
        return r;
    };
    _Float16* hW[21]; float* scv[21]; float* biv[21];
    for (int l = 0; l < nLayers; ++l) {
        hW[l]  = (_Float16*)carve((size_t)Wn[l] * 2);
        scv[l] = (float*)carve((size_t)Co[l] * 4);
        biv[l] = (float*)carve((size_t)Co[l] * 4);
    }
    _Float16* bufStem = (_Float16*)carve((size_t)32 * 64 * 112 * 112 * 2);
    _Float16* bufs[3];
    for (int i = 0; i < 3; ++i)
        bufs[i] = (_Float16*)carve((size_t)32 * 64 * 56 * 56 * 2);
    float* v512 = (float*)carve((size_t)32 * 512 * 4);
    float* v256 = (float*)carve((size_t)32 * 256 * 4);

    // ---- prep: weight f16 conversion + BN folding ----
    for (int l = 0; l < nLayers; ++l) {
        f2h_kernel<<<(Wn[l] + 255) / 256, 256, 0, stream>>>(Wsrc[l], hW[l], Wn[l]);
        bnfold_kernel<<<(Co[l] + 63) / 64, 64, 0, stream>>>(
            Ls[l], Lb[l], Lm[l], Lv[l], scv[l], biv[l], Co[l]);
    }

    // ---- stem conv + maxpool ----
    {
        int total = 32 * 64 * 112 * 112;
        stem_conv_kernel<<<(total + 255) / 256, 256, 0, stream>>>(
            X, conv1_w, scv[0], biv[0], bufStem, total);
        int ptot = 32 * 64 * 56 * 56;
        maxpool_kernel<<<(ptot + 255) / 256, 256, 0, stream>>>(bufStem, bufs[0], ptot);
    }

    // ---- residual stages via WMMA conv ----
    auto CONV = [&](const _Float16* in, int ci, int hi, int li,
                    int co, int ho, int kw, int st, int pad,
                    const _Float16* res, _Float16* out, int relu) {
        int Nt = 32 * ho * ho;
        dim3 grid((Nt + BN - 1) / BN, co / BM);
        conv_wmma_kernel<<<grid, 256, 0, stream>>>(
            in, hW[li], scv[li], biv[li], res, out,
            32, ci, hi, hi, co, ho, ho, kw, st, pad, relu);
    };

    const int Hs[5] = {56, 56, 28, 14, 7};
    const int strides[4] = {1, 2, 2, 2};
    int cI = 0;     // bufs[cI] holds current activation
    int li = 1;     // layer index (0 was stem)
    for (int s = 0; s < 4; ++s) {
        const int Hin = Hs[s], Hout = Hs[s + 1];
        const int Cin = cins[s], Cout = couts[s];
        const int st  = strides[s];
        // block 0 (projection shortcut when channels change)
        {
            int iIn = cI, iT = (cI + 1) % 3, iO = (cI + 2) % 3;
            const _Float16* resid;
            _Float16* outb;
            if (s > 0) {
                CONV(bufs[iIn], Cin, Hin, li + 2, Cout, Hout, 1, st, 0,
                     nullptr, bufs[iO], 0);
                resid = bufs[iO];
                outb  = bufs[iIn];
            } else {
                resid = bufs[iIn];
                outb  = bufs[iO];
            }
            CONV(bufs[iIn], Cin, Hin, li, Cout, Hout, 3, st, 1, nullptr, bufs[iT], 1);
            CONV(bufs[iT], Cout, Hout, li + 1, Cout, Hout, 3, 1, 1, resid, outb, 1);
            li += (s > 0) ? 3 : 2;
            cI = (s > 0) ? iIn : iO;
        }
        // block 1 (identity shortcut)
        {
            int iIn = cI, iT = (cI + 1) % 3, iO = (cI + 2) % 3;
            CONV(bufs[iIn], Cout, Hout, li,     Cout, Hout, 3, 1, 1, nullptr,  bufs[iT], 1);
            CONV(bufs[iT],  Cout, Hout, li + 1, Cout, Hout, 3, 1, 1, bufs[iIn], bufs[iO], 1);
            cI = iO;
            li += 2;
        }
    }

    // ---- head: avgpool -> fc1 -> decoder ----
    avgpool_kernel<<<(32 * 512 + 255) / 256, 256, 0, stream>>>(bufs[cI], v512);
    fc1_kernel<<<(32 * 256 + 255) / 256, 256, 0, stream>>>(v512, fc1_w, fc1_b, v256);
    decoder_kernel<<<dim3((2500 + 255) / 256, 32), 256, 0, stream>>>(
        v256, coords, dec1_w, dec1_b, dec2_w, dec2_b, (float*)d_out);
}
